// GraphHeadAttention_6725918785634
// MI455X (gfx1250) — matrix operand order repaired and checked by reading, compiled, not run on hardware
//
#include <hip/hip_runtime.h>

#define HEADS    12
#define DIM_HEAD 64
#define INNER    768
#define DIM      768
#define BATCH    8
#define SEQ      1024
#define QKV3     2304   // 3 * INNER

typedef __attribute__((ext_vector_type(16))) _Float16 v16h;
typedef __attribute__((ext_vector_type(8)))  _Float16 v8h;
typedef __attribute__((ext_vector_type(8)))  float    v8f;

static __device__ __forceinline__ v8h ld8h(const _Float16* p) {
    return *(const v8h*)p;
}

static __device__ __forceinline__ v16h pack16(v8h lo, v8h hi) {
    v16h r;
#pragma unroll
    for (int i = 0; i < 8; ++i) { r[i] = lo[i]; r[i + 8] = hi[i]; }
    return r;
}

// out[n*K + k] = (f16) in[k*Nc + n]   (cast + transpose so GEMM B-columns are contiguous)
__global__ void k_cast_transpose(const float* __restrict__ in, _Float16* __restrict__ out,
                                 int K, int Nc) {
    int idx = blockIdx.x * blockDim.x + threadIdx.x;
    if (idx >= K * Nc) return;
    int n = idx / K;
    int k = idx - n * K;
    out[idx] = (_Float16)in[(size_t)k * Nc + n];
}

// One block per token row: LayerNorm(768) -> f16
__global__ void k_layernorm(const float* __restrict__ x, const float* __restrict__ g,
                            const float* __restrict__ bb, _Float16* __restrict__ xn) {
    __shared__ float red[256];
    const int row = blockIdx.x;
    const int tid = threadIdx.x;
    const float* xr = x + (size_t)row * DIM;
    float v0 = xr[tid], v1 = xr[tid + 256], v2 = xr[tid + 512];
    red[tid] = v0 + v1 + v2;
    __syncthreads();
    for (int off = 128; off > 0; off >>= 1) {
        if (tid < off) red[tid] += red[tid + off];
        __syncthreads();
    }
    float mean = red[0] * (1.0f / DIM);
    __syncthreads();
    float d0 = v0 - mean, d1 = v1 - mean, d2 = v2 - mean;
    red[tid] = d0 * d0 + d1 * d1 + d2 * d2;
    __syncthreads();
    for (int off = 128; off > 0; off >>= 1) {
        if (tid < off) red[tid] += red[tid + off];
        __syncthreads();
    }
    float rstd = rsqrtf(red[0] * (1.0f / DIM) + 1e-5f);
    _Float16* o = xn + (size_t)row * DIM;
    o[tid]       = (_Float16)(d0 * rstd * g[tid]       + bb[tid]);
    o[tid + 256] = (_Float16)(d1 * rstd * g[tid + 256] + bb[tid + 256]);
    o[tid + 512] = (_Float16)(d2 * rstd * g[tid + 512] + bb[tid + 512]);
}

// C[M,Nt] = A[M,K] @ Bt[Nt,K]^T, f32 accumulate.
// 4 waves/block; each wave computes a 16x64 strip (4 accumulators sharing one A
// operand per k-step -> 4 WMMAs per A fetch). Block tile = 16 x 256.
// OUT32: write float + bias (final projection); else write f16.
template<bool OUT32>
__global__ void k_gemm(const _Float16* __restrict__ A, const _Float16* __restrict__ Bt,
                       const float* __restrict__ bias, void* __restrict__ Cv,
                       int K, int Nt) {
    const int wave = threadIdx.x >> 5;
    const int lane = threadIdx.x & 31;
    const int half = lane >> 4;
    const int l15  = lane & 15;
    const int mbase = blockIdx.x * 16;
    const int nb    = blockIdx.y * 256 + wave * 64;

    // A operand: row = lane&15, 16 contiguous K-halves at half*16
    const _Float16* arow = A + (size_t)(mbase + l15) * K + half * 16;
    // B operand tile t: col = lane&15, 16 contiguous K-halves at half*16
    const _Float16* brow0 = Bt + (size_t)(nb + 0  + l15) * K + half * 16;
    const _Float16* brow1 = Bt + (size_t)(nb + 16 + l15) * K + half * 16;
    const _Float16* brow2 = Bt + (size_t)(nb + 32 + l15) * K + half * 16;
    const _Float16* brow3 = Bt + (size_t)(nb + 48 + l15) * K + half * 16;

    v8f acc0 = {}, acc1 = {}, acc2 = {}, acc3 = {};
    for (int kk = 0; kk < K; kk += 32) {
        __builtin_prefetch(brow0 + kk + 512, 0, 1);   // global_prefetch_b8, 16 k-steps (512 halves) ahead
        v16h a  = pack16(ld8h(arow  + kk), ld8h(arow  + kk + 8));
        v16h b0 = pack16(ld8h(brow0 + kk), ld8h(brow0 + kk + 8));
        v16h b1 = pack16(ld8h(brow1 + kk), ld8h(brow1 + kk + 8));
        v16h b2 = pack16(ld8h(brow2 + kk), ld8h(brow2 + kk + 8));
        v16h b3 = pack16(ld8h(brow3 + kk), ld8h(brow3 + kk + 8));
        acc0 = __builtin_amdgcn_wmma_f32_16x16x32_f16(false, a, false, b0, (short)0, acc0, false, false);
        acc1 = __builtin_amdgcn_wmma_f32_16x16x32_f16(false, a, false, b1, (short)0, acc1, false, false);
        acc2 = __builtin_amdgcn_wmma_f32_16x16x32_f16(false, a, false, b2, (short)0, acc2, false, false);
        acc3 = __builtin_amdgcn_wmma_f32_16x16x32_f16(false, a, false, b3, (short)0, acc3, false, false);
    }

    const int mrow = mbase + half * 8;     // D layout: lane>>4 selects M-half, VGPR r = row
    if (OUT32) {
        float* C = (float*)Cv;
#pragma unroll
        for (int t = 0; t < 4; ++t) {
            v8f a4 = (t == 0) ? acc0 : (t == 1) ? acc1 : (t == 2) ? acc2 : acc3;
            float bv = bias[nb + t * 16 + l15];
            float* crow = C + (size_t)mrow * Nt + nb + t * 16 + l15;
#pragma unroll
            for (int r = 0; r < 8; ++r) crow[(size_t)r * Nt] = a4[r] + bv;
        }
    } else {
        _Float16* C = (_Float16*)Cv;
#pragma unroll
        for (int t = 0; t < 4; ++t) {
            v8f a4 = (t == 0) ? acc0 : (t == 1) ? acc1 : (t == 2) ? acc2 : acc3;
            _Float16* crow = C + (size_t)mrow * Nt + nb + t * 16 + l15;
#pragma unroll
            for (int r = 0; r < 8; ++r) crow[(size_t)r * Nt] = (_Float16)a4[r];
        }
    }
}

// Attention with top-k (4/8/16) masked-softmax mixing.
// Block = 128 thr (4 waves) handles 16 queries of one (b,h).
__global__ void k_attention(const _Float16* __restrict__ qkv, const float* __restrict__ sw,
                            _Float16* __restrict__ attn_out) {
    __shared__ float sc[16][64];   // score chunk: 16 queries x 64 keys
    __shared__ float cf[16][16];   // per-query mixed softmax coefficients
    __shared__ int   ixs[16][16];  // per-query top-16 key indices

    const int qb  = blockIdx.x * 16;
    const int h   = blockIdx.y;
    const int b   = blockIdx.z;
    const int tid = threadIdx.x;
    const int wave = tid >> 5;
    const int lane = tid & 31;
    const int half = lane >> 4;
    const int l15  = lane & 15;
    const size_t rs = QKV3;

    // Q tile (16x64) operands for the two chained K=32 WMMAs; reused for all key tiles.
    const _Float16* qrow = qkv + ((size_t)b * SEQ + qb + l15) * rs
                         + (size_t)h * DIM_HEAD + half * 16;
    v16h aq0 = pack16(ld8h(qrow + 0),  ld8h(qrow + 8));
    v16h aq1 = pack16(ld8h(qrow + 32), ld8h(qrow + 40));

    float val[16];
    int   vid[16];
#pragma unroll
    for (int i = 0; i < 16; ++i) { val[i] = -3.0e38f; vid[i] = 0; }

    const _Float16* kbp = qkv + (size_t)b * SEQ * rs + INNER
                        + (size_t)h * DIM_HEAD + half * 16;

    for (int it = 0; it < 16; ++it) {
        // ---- 4 waves compute 64 keys' scores via WMMA (full EXEC, no divergence) ----
        const int key0 = it * 64 + wave * 16;
        const _Float16* kr = kbp + (size_t)(key0 + l15) * rs;
        v16h b0 = pack16(ld8h(kr + 0),  ld8h(kr + 8));    // dims half*16 + 0..15
        v16h b1 = pack16(ld8h(kr + 32), ld8h(kr + 40));   // dims 32 + half*16 + 0..15
        v8f acc = {};
        acc = __builtin_amdgcn_wmma_f32_16x16x32_f16(false, aq0, false, b0,
                                                     (short)0, acc, false, false);
        acc = __builtin_amdgcn_wmma_f32_16x16x32_f16(false, aq1, false, b1,
                                                     (short)0, acc, false, false);
#pragma unroll
        for (int r = 0; r < 8; ++r)
            sc[half * 8 + r][wave * 16 + l15] = acc[r] * 0.125f;   // * DIM_HEAD^-0.5
        __syncthreads();

        // ---- 16 threads merge 64 candidates into register-resident sorted top-16 ----
        if (tid < 16) {
#pragma unroll 1
            for (int jj = 0; jj < 64; ++jj) {
                float s = sc[tid][jj];
                if (s > val[15]) {
                    int key = it * 64 + jj;
#pragma unroll
                    for (int p = 15; p >= 1; --p) {
                        bool gt  = s > val[p];
                        bool gtp = s > val[p - 1];
                        float nv = gt ? (gtp ? val[p - 1] : s)   : val[p];
                        int   ni = gt ? (gtp ? vid[p - 1] : key) : vid[p];
                        val[p] = nv; vid[p] = ni;
                    }
                    if (s > val[0]) { val[0] = s; vid[0] = key; }
                }
            }
        }
        __syncthreads();
    }

    // ---- prefix softmax over sorted top-4/8/16, mixed with softmax(scale_weights) ----
    if (tid < 16) {
        float m = val[0];
        float e[16], S4 = 0.0f, S8 = 0.0f, S16 = 0.0f;
#pragma unroll
        for (int p = 0; p < 16; ++p) {
            e[p] = __expf(val[p] - m);
            if (p < 4) S4 += e[p];
            if (p < 8) S8 += e[p];
            S16 += e[p];
        }
        float w0 = sw[0], w1 = sw[1], w2 = sw[2];
        float wm = fmaxf(w0, fmaxf(w1, w2));
        float x0 = __expf(w0 - wm), x1 = __expf(w1 - wm), x2 = __expf(w2 - wm);
        float inv = 1.0f / (x0 + x1 + x2);
        float g0 = x0 * inv / S4, g1 = x1 * inv / S8, g2 = x2 * inv / S16;
#pragma unroll
        for (int p = 0; p < 16; ++p) {
            cf[tid][p]  = e[p] * (g2 + (p < 8 ? g1 : 0.0f) + (p < 4 ? g0 : 0.0f));
            ixs[tid][p] = vid[p];
        }
    }
    __syncthreads();

    // ---- sparse attn @ V: 8 threads per query, 8 dims per thread, 16-row gather ----
    {
        const int q  = tid >> 3;
        const int d0 = (tid & 7) * 8;
        float acc[8];
#pragma unroll
        for (int i = 0; i < 8; ++i) acc[i] = 0.0f;
        const _Float16* vbase = qkv + (size_t)b * SEQ * rs + 2 * INNER
                              + (size_t)h * DIM_HEAD + d0;
#pragma unroll 1
        for (int j = 0; j < 16; ++j) {
            float c = cf[q][j];
            v8h vv = ld8h(vbase + (size_t)ixs[q][j] * rs);
#pragma unroll
            for (int i = 0; i < 8; ++i) acc[i] += c * (float)vv[i];
        }
        _Float16* op = attn_out + ((size_t)b * SEQ + qb + q) * INNER
                     + (size_t)h * DIM_HEAD + d0;
#pragma unroll
        for (int i = 0; i < 8; ++i) op[i] = (_Float16)acc[i];
    }
}

extern "C" void kernel_launch(void* const* d_in, const int* in_sizes, int n_in,
                              void* d_out, int out_size, void* d_ws, size_t ws_size,
                              hipStream_t stream) {
    (void)in_sizes; (void)n_in; (void)out_size; (void)ws_size;
    const float* x    = (const float*)d_in[0];
    const float* ln_g = (const float*)d_in[1];
    const float* ln_b = (const float*)d_in[2];
    const float* wqkv = (const float*)d_in[3];
    const float* wout = (const float*)d_in[4];
    const float* bout = (const float*)d_in[5];
    const float* swts = (const float*)d_in[6];
    float* out = (float*)d_out;

    char* ws = (char*)d_ws;
    _Float16* wqkvT = (_Float16*)ws;  ws += (size_t)QKV3 * DIM * 2;        // [2304,768]
    _Float16* woutT = (_Float16*)ws;  ws += (size_t)DIM * DIM * 2;         // [768,768]
    _Float16* xn    = (_Float16*)ws;  ws += (size_t)BATCH * SEQ * DIM * 2; // [8192,768]
    _Float16* qkvh  = (_Float16*)ws;  ws += (size_t)BATCH * SEQ * QKV3 * 2;// [8192,2304]
    _Float16* attnh = (_Float16*)ws;                                       // [8192,768]

    k_cast_transpose<<<(DIM * QKV3 + 255) / 256, 256, 0, stream>>>(wqkv, wqkvT, DIM, QKV3);
    k_cast_transpose<<<(DIM * DIM  + 255) / 256, 256, 0, stream>>>(wout, woutT, DIM, DIM);
    k_layernorm<<<BATCH * SEQ, 256, 0, stream>>>(x, ln_g, ln_b, xn);
    // QKV GEMM: 16x256 block tiles; 2304 = 9 * 256
    k_gemm<false><<<dim3(BATCH * SEQ / 16, QKV3 / 256), 128, 0, stream>>>(
        xn, wqkvT, nullptr, (void*)qkvh, DIM, QKV3);
    k_attention<<<dim3(SEQ / 16, HEADS, BATCH), 128, 0, stream>>>(qkvh, swts, attnh);
    // Output projection: 768 = 3 * 256
    k_gemm<true><<<dim3(BATCH * SEQ / 16, DIM / 256), 128, 0, stream>>>(
        attnh, woutT, bout, (void*)out, DIM, DIM);
}
